// GGNN_Encoder_1159641169925
// MI455X (gfx1250) — compile-verified, matrix operand order checked by reading, not run on hardware
//
#include <hip/hip_runtime.h>
#include <hip/hip_bf16.h>

// Problem constants (from the reference)
#define NN 10000
#define HH 512
#define TT 4
#define EE 40000
#define LL 2
#define EPT32 (EE / 32)        // 1250 edge tiles (32 edges) per type
#define NT16  (NN / 16)        // 625 node tiles (16 rows)
#define NT32  ((NN + 31) / 32) // 313 node tiles (32 rows, last partial)
#define HT16  (HH / 16)        // 32 column tiles
#define LDSTR 520              // padded LDS row stride (bf16 elems) to dodge bank conflicts

typedef __bf16 bf16_t;
typedef __attribute__((ext_vector_type(16))) __bf16 v16bf;
typedef __attribute__((ext_vector_type(8)))  __bf16 v8bf;
typedef __attribute__((ext_vector_type(8)))  float   v8f;

// Build a 16-element bf16 WMMA fragment from two contiguous 8-element (16B) runs.
__device__ __forceinline__ v16bf make_frag(const bf16_t* __restrict__ p0,
                                           const bf16_t* __restrict__ p1) {
    v8bf lo = *(const v8bf*)p0;
    v8bf hi = *(const v8bf*)p1;
    v16bf r;
#pragma unroll
    for (int i = 0; i < 8; ++i) { r[i] = lo[i]; r[i + 8] = hi[i]; }
    return r;
}

__device__ __forceinline__ v8f wmma_bf16(v16bf a, v16bf b, v8f c) {
    return __builtin_amdgcn_wmma_f32_16x16x32_bf16(false, a, false, b, (short)0, c,
                                                   false, false);
}

// ---------- utility kernels ----------
__global__ void k_zero_f32(float* __restrict__ p, int n) {
    int i = blockIdx.x * blockDim.x + threadIdx.x;
    if (i < n) p[i] = 0.0f;
}

__global__ void k_f2bf(const float* __restrict__ in, bf16_t* __restrict__ out, int n) {
    int i = blockIdx.x * blockDim.x + threadIdx.x;
    if (i < n) out[i] = (bf16_t)in[i];
}

// monotone float->uint mapping for atomicMax-based float max
__global__ void k_init_out(unsigned* __restrict__ o) {
    int i = blockIdx.x * blockDim.x + threadIdx.x;
    if (i < HH) o[i] = 0x007FFFFFu;    // mapped(-inf)
}
__global__ void k_final_out(float* __restrict__ o) {
    int i = blockIdx.x * blockDim.x + threadIdx.x;
    if (i < HH) {
        unsigned m = ((unsigned*)o)[i];
        unsigned u = (m & 0x80000000u) ? (m & 0x7FFFFFFFu) : ~m;
        o[i] = __uint_as_float(u);
    }
}

// ---------- edge-message GEMM + scatter-add ----------
// One wave per (edge type, 32-edge tile): msgs(32x512) = src_states(32x512) @ W^T.
// Two 16-row subtiles share every B fragment (2 WMMAs per B load).
__global__ __launch_bounds__(32)
void k_msg(const bf16_t* __restrict__ h_bf,
           const bf16_t* __restrict__ Wm,     // (T,H,H), this layer
           const float*  __restrict__ bm,     // (T,H),   this layer
           const int*    __restrict__ edges,  // (T,E,2)
           float*        __restrict__ inc)    // (N,H)
{
    __shared__ bf16_t As[32 * LDSTR];
    __shared__ int    tgts[32];

    const int blk  = blockIdx.x;
    const int t    = blk / EPT32;
    const int e0   = (blk % EPT32) * 32;
    const int lane = threadIdx.x;

    // stage target ids + gather 32 source rows into LDS (one 1KB row per lane)
    tgts[lane] = edges[((size_t)t * EE + e0 + lane) * 2 + 1];
    {
        const int src = edges[((size_t)t * EE + e0 + lane) * 2 + 0];
        const bf16_t* sp = h_bf + (size_t)src * HH;
        bf16_t*       dp = As + lane * LDSTR;
#pragma unroll 4
        for (int i = 0; i < HH / 8; ++i) ((v8bf*)dp)[i] = ((const v8bf*)sp)[i];
    }
    __syncthreads();

    const int koff  = (lane < 16) ? 0 : 8;
    const int m     = lane & 15;
    const int n     = lane & 15;
    const int mbase = (lane >= 16) ? 8 : 0;
    const bf16_t* Wt = Wm + (size_t)t * HH * HH;

    for (int n0 = 0; n0 < HT16; ++n0) {
        const bf16_t* wrow = Wt + (size_t)(n0 * 16 + n) * HH;
        if (n0 + 1 < HT16)
            __builtin_prefetch(Wt + (size_t)((n0 + 1) * 16 + n) * HH, 0, 0);
        v8f acc1 = {}, acc2 = {};
#pragma unroll 4
        for (int k0 = 0; k0 < HH; k0 += 32) {
            v16bf b  = make_frag(wrow + k0 + koff, wrow + k0 + 16 + koff);
            v16bf a1 = make_frag(&As[m * LDSTR + k0 + koff],
                                 &As[m * LDSTR + k0 + 16 + koff]);
            v16bf a2 = make_frag(&As[(16 + m) * LDSTR + k0 + koff],
                                 &As[(16 + m) * LDSTR + k0 + 16 + koff]);
            acc1 = wmma_bf16(a1, b, acc1);
            acc2 = wmma_bf16(a2, b, acc2);
        }
        const float bias = bm[t * HH + n0 * 16 + n];
        const int   colg = n0 * 16 + n;
#pragma unroll
        for (int r = 0; r < 8; ++r) {
            atomicAdd(&inc[(size_t)tgts[r + mbase] * HH + colg], acc1[r] + bias);
            atomicAdd(&inc[(size_t)tgts[16 + r + mbase] * HH + colg], acc2[r] + bias);
        }
    }
}

// ---------- fused GRU: 6 gate GEMM tiles x 2 row subtiles + pointwise ----------
// One wave per (32-node tile, 16-col tile); every B fragment feeds 2 WMMAs.
__global__ __launch_bounds__(32)
void k_gru(const float*  __restrict__ inc,
           const float*  __restrict__ h_cur,
           const bf16_t* __restrict__ h_bf,
           const bf16_t* __restrict__ Wih,    // (3H,H) this layer
           const bf16_t* __restrict__ Whh,    // (3H,H) this layer
           const float*  __restrict__ bih,    // (3H)
           const float*  __restrict__ bhh,    // (3H)
           float*        __restrict__ h_nxt)
{
    __shared__ bf16_t Xs[32 * LDSTR];         // incoming tile, bf16

    const int blk  = blockIdx.x;
    const int i0   = (blk / HT16) * 32;
    const int j    = (blk % HT16);
    const int lane = threadIdx.x;

    // stage incoming rows, f32 -> bf16, one row per lane (clamped in-bounds)
    {
        int rg = i0 + lane; if (rg > NN - 1) rg = NN - 1;
        const float* sp = inc + (size_t)rg * HH;
        bf16_t*      dp = Xs + lane * LDSTR;
#pragma unroll 4
        for (int i = 0; i < HH; i += 8) {
            v8bf v;
#pragma unroll
            for (int q = 0; q < 8; ++q) v[q] = (bf16_t)sp[i + q];
            *(v8bf*)(dp + i) = v;
        }
    }
    __syncthreads();

    const int koff  = (lane < 16) ? 0 : 8;
    const int m     = lane & 15;
    const int n     = lane & 15;
    const int mbase = (lane >= 16) ? 8 : 0;
    const int col   = j * 16 + n;

    const int r1 = i0 + m;
    int r2 = i0 + 16 + m; if (r2 > NN - 1) r2 = NN - 1;   // clamp last partial tile
    const bf16_t* hrow1 = h_bf + (size_t)r1 * HH;
    const bf16_t* hrow2 = h_bf + (size_t)r2 * HH;

    const bf16_t* w[6] = {
        Wih + (size_t)(0 * HH + col) * HH, Wih + (size_t)(1 * HH + col) * HH,
        Wih + (size_t)(2 * HH + col) * HH,
        Whh + (size_t)(0 * HH + col) * HH, Whh + (size_t)(1 * HH + col) * HH,
        Whh + (size_t)(2 * HH + col) * HH };

    v8f acc1[6] = {}, acc2[6] = {};

#pragma unroll 2
    for (int k0 = 0; k0 < HH; k0 += 32) {
        v16bf ax1 = make_frag(&Xs[m * LDSTR + k0 + koff],
                              &Xs[m * LDSTR + k0 + 16 + koff]);
        v16bf ax2 = make_frag(&Xs[(16 + m) * LDSTR + k0 + koff],
                              &Xs[(16 + m) * LDSTR + k0 + 16 + koff]);
        v16bf ah1 = make_frag(hrow1 + k0 + koff, hrow1 + k0 + 16 + koff);
        v16bf ah2 = make_frag(hrow2 + k0 + koff, hrow2 + k0 + 16 + koff);
#pragma unroll
        for (int g = 0; g < 3; ++g) {
            v16bf b = make_frag(w[g] + k0 + koff, w[g] + k0 + 16 + koff);
            acc1[g] = wmma_bf16(ax1, b, acc1[g]);
            acc2[g] = wmma_bf16(ax2, b, acc2[g]);
        }
#pragma unroll
        for (int g = 0; g < 3; ++g) {
            v16bf b = make_frag(w[3 + g] + k0 + koff, w[3 + g] + k0 + 16 + koff);
            acc1[3 + g] = wmma_bf16(ah1, b, acc1[3 + g]);
            acc2[3 + g] = wmma_bf16(ah2, b, acc2[3 + g]);
        }
    }

    const float bi0 = bih[col], bi1 = bih[HH + col], bi2 = bih[2 * HH + col];
    const float bh0 = bhh[col], bh1 = bhh[HH + col], bh2 = bhh[2 * HH + col];

#pragma unroll
    for (int sub = 0; sub < 2; ++sub) {
        const v8f* A = sub ? acc2 : acc1;
#pragma unroll
        for (int r = 0; r < 8; ++r) {
            const int node = i0 + sub * 16 + r + mbase;
            if (node < NN) {
                const float xr = A[0][r] + bi0, xz = A[1][r] + bi1, xn = A[2][r] + bi2;
                const float yr = A[3][r] + bh0, yz = A[4][r] + bh1, yn = A[5][r] + bh2;
                const float rg = 1.0f / (1.0f + __expf(-(xr + yr)));
                const float zg = 1.0f / (1.0f + __expf(-(xz + yz)));
                const float ng = tanhf(xn + rg * yn);
                const float hv = h_cur[(size_t)node * HH + col];
                h_nxt[(size_t)node * HH + col] = (1.0f - zg) * ng + zg * hv;
            }
        }
    }
}

// ---------- FC + graph max (mapped-uint atomicMax) ----------
__global__ __launch_bounds__(32)
void k_fc_max(const bf16_t* __restrict__ h_bf,
              const bf16_t* __restrict__ fcW,   // (H,H)
              const float*  __restrict__ fcb,   // (H)
              unsigned*     __restrict__ out_u) // (H) mapped
{
    const int blk  = blockIdx.x;
    const int i0   = (blk / HT16) * 16;
    const int j    = (blk % HT16);
    const int lane = threadIdx.x;

    const int koff = (lane < 16) ? 0 : 8;
    const int m    = lane & 15;
    const int n    = lane & 15;
    const int col  = j * 16 + n;

    const bf16_t* hrow = h_bf + (size_t)(i0 + m) * HH;
    const bf16_t* wrow = fcW + (size_t)col * HH;

    v8f acc = {};
#pragma unroll 4
    for (int k0 = 0; k0 < HH; k0 += 32) {
        v16bf a = make_frag(hrow + k0 + koff, hrow + k0 + 16 + koff);
        v16bf b = make_frag(wrow + k0 + koff, wrow + k0 + 16 + koff);
        acc = wmma_bf16(a, b, acc);
    }
    const float bias = fcb[col];
    unsigned best = 0;
#pragma unroll
    for (int r = 0; r < 8; ++r) {
        const float v = acc[r] + bias;
        const unsigned u = __float_as_uint(v);
        const unsigned mm = (u & 0x80000000u) ? ~u : (u | 0x80000000u);
        best = best > mm ? best : mm;
    }
    atomicMax(&out_u[col], best);
}

// ---------- host orchestration ----------
extern "C" void kernel_launch(void* const* d_in, const int* in_sizes, int n_in,
                              void* d_out, int out_size, void* d_ws, size_t ws_size,
                              hipStream_t stream) {
    (void)in_sizes; (void)n_in; (void)out_size; (void)ws_size;

    const float* x     = (const float*)d_in[0];
    const int*   edges = (const int*)d_in[1];
    const float* W_msg = (const float*)d_in[2];  // (L,T,H,H)
    const float* b_msg = (const float*)d_in[3];  // (L,T,H)
    const float* W_ih  = (const float*)d_in[4];  // (L,3H,H)
    const float* W_hh  = (const float*)d_in[5];  // (L,3H,H)
    const float* b_ih  = (const float*)d_in[6];  // (L,3H)
    const float* b_hh  = (const float*)d_in[7];  // (L,3H)
    const float* fc_W  = (const float*)d_in[8];  // (H,H)
    const float* fc_b  = (const float*)d_in[9];  // (H)

    char* ws = (char*)d_ws;
    size_t off = 0;
    auto alloc = [&](size_t bytes) -> void* {
        void* p = ws + off;
        off += (bytes + 255) & ~(size_t)255;
        return p;
    };
    bf16_t* h_bf    = (bf16_t*)alloc((size_t)NN * HH * 2);
    float*  inc     = (float*) alloc((size_t)NN * HH * 4);
    float*  hA      = (float*) alloc((size_t)NN * HH * 4);
    float*  hB      = (float*) alloc((size_t)NN * HH * 4);
    bf16_t* Wmsg_bf = (bf16_t*)alloc((size_t)LL * TT * HH * HH * 2);
    bf16_t* Wih_bf  = (bf16_t*)alloc((size_t)LL * 3 * HH * HH * 2);
    bf16_t* Whh_bf  = (bf16_t*)alloc((size_t)LL * 3 * HH * HH * 2);
    bf16_t* fcW_bf  = (bf16_t*)alloc((size_t)HH * HH * 2);

    const int NH = NN * HH;
    auto gsz = [](int n) { return (n + 255) / 256; };

    // weight conversion (runs every call for determinism; cheap)
    k_f2bf<<<gsz(LL * TT * HH * HH), 256, 0, stream>>>(W_msg, Wmsg_bf, LL * TT * HH * HH);
    k_f2bf<<<gsz(LL * 3 * HH * HH), 256, 0, stream>>>(W_ih, Wih_bf, LL * 3 * HH * HH);
    k_f2bf<<<gsz(LL * 3 * HH * HH), 256, 0, stream>>>(W_hh, Whh_bf, LL * 3 * HH * HH);
    k_f2bf<<<gsz(HH * HH), 256, 0, stream>>>(fc_W, fcW_bf, HH * HH);

    hipMemcpyAsync(hA, x, (size_t)NH * 4, hipMemcpyDeviceToDevice, stream);

    float* hc = hA;
    float* hn = hB;
    const int TIMESTEPS[LL] = {2, 2};
    for (int layer = 0; layer < LL; ++layer) {
        for (int step = 0; step < TIMESTEPS[layer]; ++step) {
            k_zero_f32<<<gsz(NH), 256, 0, stream>>>(inc, NH);
            k_f2bf<<<gsz(NH), 256, 0, stream>>>(hc, h_bf, NH);
            k_msg<<<TT * EPT32, 32, 0, stream>>>(
                h_bf,
                Wmsg_bf + (size_t)layer * TT * HH * HH,
                b_msg + (size_t)layer * TT * HH,
                edges, inc);
            k_gru<<<NT32 * HT16, 32, 0, stream>>>(
                inc, hc, h_bf,
                Wih_bf + (size_t)layer * 3 * HH * HH,
                Whh_bf + (size_t)layer * 3 * HH * HH,
                b_ih + (size_t)layer * 3 * HH,
                b_hh + (size_t)layer * 3 * HH,
                hn);
            float* tmp = hc; hc = hn; hn = tmp;
        }
    }

    k_f2bf<<<gsz(NH), 256, 0, stream>>>(hc, h_bf, NH);
    k_init_out<<<2, 256, 0, stream>>>((unsigned*)d_out);
    k_fc_max<<<NT16 * HT16, 32, 0, stream>>>(h_bf, fcW_bf, fc_b, (unsigned*)d_out);
    k_final_out<<<2, 256, 0, stream>>>((float*)d_out);
}
